// CrossAttention_7370163880507
// MI455X (gfx1250) — matrix operand order repaired and checked by reading, not compiled, not run on hardware
//
#include <hip/hip_runtime.h>
#include <hip/hip_bf16.h>
#include <math.h>

typedef __bf16 bf16;
typedef __attribute__((ext_vector_type(16))) __bf16 v16bf;
typedef __attribute__((ext_vector_type(8)))  __bf16 v8bf;
typedef __attribute__((ext_vector_type(8)))  float  v8f;

// D = 128, Q = 2048 (per view), N = 4 views, HEADS = 4, DH = 32, HD = 128.

__device__ inline v8f wmma_bf16(v16bf a, v16bf b, v8f c) {
    return __builtin_amdgcn_wmma_f32_16x16x32_bf16(
        /*neg_a=*/false, a, /*neg_b=*/false, b,
        /*c_mod=*/(short)0, c, /*reuse_a=*/false, /*reuse_b=*/false);
}

// reductions within 16-lane halves (C-layout row groups); xor masks < 16 stay in-half
__device__ inline float grp_max(float v) {
    #pragma unroll
    for (int m = 1; m < 16; m <<= 1) v = fmaxf(v, __shfl_xor(v, m, 32));
    return v;
}
__device__ inline float grp_sum(float v) {
    #pragma unroll
    for (int m = 1; m < 16; m <<= 1) v += __shfl_xor(v, m, 32);
    return v;
}

// ---------------------------------------------------------------------------
// K1: fused LayerNorm(d=128) + Linear(128->128) for q/k/v.
// Input x layout: [N=4][D=128][Q=2048]; logical row = n*2048 + qpos.
// Output bf16: row-major [8192][128], or (TRANSPOSE=1) [128][8192] for V.
// For q, scale folds (1/sqrt(dh)) * log2(e) so attention can use exp2.
// ---------------------------------------------------------------------------
template <int TRANSPOSE>
__global__ void __launch_bounds__(256)
ln_proj_kernel(const float* __restrict__ x, const float* __restrict__ g,
               const float* __restrict__ beta, const float* __restrict__ W,
               const float* __restrict__ bias, bf16* __restrict__ out,
               float scale) {
    __shared__ float xs[16][128];
    __shared__ __align__(32) bf16 xn[16][128];
    __shared__ __align__(32) bf16 wT[128][128];   // wT[col][k] = W[k][col]
    __shared__ float ssum[16], ssq[16];

    const int tid = threadIdx.x;
    const int rowbase = blockIdx.x * 16;

    for (int idx = tid; idx < 128 * 128; idx += 256) {
        int k = idx >> 7, c = idx & 127;
        wT[c][k] = (bf16)W[idx];
    }
    for (int idx = tid; idx < 16 * 128; idx += 256) {
        int r = idx >> 7, d = idx & 127;
        int row = rowbase + r;
        xs[r][d] = x[(size_t)(row >> 11) * (128 * 2048) + (size_t)d * 2048 + (row & 2047)];
    }
    if (tid < 16) { ssum[tid] = 0.f; ssq[tid] = 0.f; }
    __syncthreads();

    const int r = tid >> 4, j = tid & 15;
    {
        float s = 0.f, s2 = 0.f;
        #pragma unroll
        for (int e = 0; e < 8; e++) {
            float v = xs[r][j * 8 + e];
            s += v; s2 += v * v;
        }
        atomicAdd(&ssum[r], s);
        atomicAdd(&ssq[r], s2);
    }
    __syncthreads();
    {
        float mean = ssum[r] * (1.0f / 128.0f);
        float var  = ssq[r] * (1.0f / 128.0f) - mean * mean;
        float rstd = rsqrtf(var + 1e-5f);
        #pragma unroll
        for (int e = 0; e < 8; e++) {
            int d = j * 8 + e;
            xn[r][d] = (bf16)((xs[r][d] - mean) * rstd * g[d] + beta[d]);
        }
    }
    __syncthreads();

    const int wave = tid >> 5, lane = tid & 31;
    const int lo = lane & 15, hi = lane >> 4;
    const int colbase = wave * 16;

    v8f acc = {};
    #pragma unroll
    for (int kk = 0; kk < 128; kk += 32) {
        const bf16* ap = &xn[lo][kk + hi * 16];
        v8bf alo = *(const v8bf*)ap;
        v8bf ahi = *(const v8bf*)(ap + 8);
        v16bf a;
        #pragma unroll
        for (int i = 0; i < 8; i++) { a[i] = alo[i]; a[i + 8] = ahi[i]; }
        v16bf b = *(const v16bf*)&wT[colbase + lo][kk + hi * 16];
        acc = wmma_bf16(a, b, acc);
    }
    const int col = colbase + lo;
    #pragma unroll
    for (int rr = 0; rr < 8; rr++) {
        int row = rowbase + rr + 8 * hi;
        float v = (acc[rr] + bias[col]) * scale;
        if (TRANSPOSE) out[(size_t)col * 8192 + row] = (bf16)v;
        else           out[(size_t)row * 128 + col]  = (bf16)v;
    }
}

// ---------------------------------------------------------------------------
// K2: attention. One wave owns one (head, 16-query tile). Per view n, the
// query vectors come from view n (per-view QK); keys iterate in blocks of 64
// (4 score WMMAs + 4 P@V WMMAs per round) with online (flash) softmax across
// all 4 views * 2048 keys. Scores carry a folded log2(e) factor so softmax
// uses native base-2 exp.
// qp/kp: bf16 [8192][128]. vpT: bf16 [128][8192].
// Output a: bf16 [2048][128] = a[qpos][head*32+dh].
// ---------------------------------------------------------------------------
__global__ void __launch_bounds__(128)
attn_kernel(const bf16* __restrict__ qp, const bf16* __restrict__ kp,
            const bf16* __restrict__ vpT, bf16* __restrict__ aout) {
    __shared__ __align__(32) bf16 Pbuf[4][16][64];

    const int tid = threadIdx.x;
    const int wave = tid >> 5, lane = tid & 31;
    const int lo = lane & 15, hi = lane >> 4;
    const int gw = blockIdx.x * 4 + wave;       // 512 waves total
    const int head  = gw >> 7;                  // 4 heads
    const int qbase = (gw & 127) << 4;          // 128 query tiles of 16

    float rm[8], rs[8];
    v8f acc0 = {}, acc1 = {};
    #pragma unroll
    for (int r = 0; r < 8; r++) { rm[r] = -1e30f; rs[r] = 0.f; }

    for (int n = 0; n < 4; n++) {
        // A operand: 16 queries x 32 (=dh), per the 16-bit A layout
        const bf16* qptr = qp + (size_t)(n * 2048 + qbase + lo) * 128 + head * 32 + hi * 16;
        v8bf qlo = *(const v8bf*)qptr;
        v8bf qhi = *(const v8bf*)(qptr + 8);
        v16bf qa;
        #pragma unroll
        for (int i = 0; i < 8; i++) { qa[i] = qlo[i]; qa[i + 8] = qhi[i]; }

        for (int kb = 0; kb < 32; kb++) {
            const int kbase = kb * 64;
            // B operands for scores: k^T (dh x keys); lane = key, contiguous dh chunk
            const bf16* kptr = kp + (size_t)(n * 2048 + kbase + lo) * 128 + head * 32 + hi * 16;
            v16bf kb0 = *(const v16bf*)kptr;
            v16bf kb1 = *(const v16bf*)(kptr + 16 * 128);
            v16bf kb2 = *(const v16bf*)(kptr + 32 * 128);
            v16bf kb3 = *(const v16bf*)(kptr + 48 * 128);
            v8f z = {};
            v8f s0 = wmma_bf16(qa, kb0, z);
            v8f s1 = wmma_bf16(qa, kb1, z);
            v8f s2 = wmma_bf16(qa, kb2, z);
            v8f s3 = wmma_bf16(qa, kb3, z);

            #pragma unroll
            for (int r = 0; r < 8; r++) {
                float tmax = fmaxf(fmaxf(s0[r], s1[r]), fmaxf(s2[r], s3[r]));
                tmax = grp_max(tmax);
                float nm = fmaxf(rm[r], tmax);
                float corr = exp2f(rm[r] - nm);
                rm[r] = nm;
                float p0 = exp2f(s0[r] - nm);
                float p1 = exp2f(s1[r] - nm);
                float p2 = exp2f(s2[r] - nm);
                float p3 = exp2f(s3[r] - nm);
                rs[r] = rs[r] * corr + grp_sum((p0 + p1) + (p2 + p3));
                acc0[r] *= corr;
                acc1[r] *= corr;
                bf16* pr = &Pbuf[wave][r + 8 * hi][lo];
                pr[0]  = (bf16)p0;
                pr[16] = (bf16)p1;
                pr[32] = (bf16)p2;
                pr[48] = (bf16)p3;
            }
            // same-wave LDS is in-order; fence stops compiler reordering
            asm volatile("s_wait_dscnt 0" ::: "memory");

            #pragma unroll
            for (int t = 0; t < 2; t++) {
                // re-read P tile (16x32 bf16) in A layout
                const bf16* pp = &Pbuf[wave][lo][t * 32 + hi * 16];
                v8bf plo = *(const v8bf*)pp;
                v8bf phi = *(const v8bf*)(pp + 8);
                v16bf pa;
                #pragma unroll
                for (int i = 0; i < 8; i++) { pa[i] = plo[i]; pa[i + 8] = phi[i]; }

                // V B-operand from transposed vpT: lane = dh, contiguous 16 keys
                const bf16* vptr = vpT + (size_t)(head * 32 + lo) * 8192
                                 + n * 2048 + kbase + t * 32 + hi * 16;
                v16bf vb0 = *(const v16bf*)vptr;
                v16bf vb1 = *(const v16bf*)(vptr + (size_t)16 * 8192);
                acc0 = wmma_bf16(pa, vb0, acc0);
                acc1 = wmma_bf16(pa, vb1, acc1);
            }
            asm volatile("" ::: "memory");
        }
    }
    #pragma unroll
    for (int r = 0; r < 8; r++) {
        float inv = 1.0f / rs[r];
        size_t row = (size_t)(qbase + r + 8 * hi);
        aout[row * 128 + head * 32 + lo]      = (bf16)(acc0[r] * inv);
        aout[row * 128 + head * 32 + 16 + lo] = (bf16)(acc1[r] * inv);
    }
}

// ---------------------------------------------------------------------------
// K3: z1 = a @ w_o + b_o + skip ; z2 = LN_pre(z1); store z2 as bf16 + f32.
// ---------------------------------------------------------------------------
__global__ void __launch_bounds__(256)
out_proj_kernel(const bf16* __restrict__ A, const float* __restrict__ W,
                const float* __restrict__ bias, const float* __restrict__ skip,
                const float* __restrict__ g, const float* __restrict__ beta,
                bf16* __restrict__ z2b, float* __restrict__ z2f) {
    __shared__ __align__(32) bf16 wT[128][128];
    __shared__ float zbuf[16][128];
    __shared__ float ssum[16], ssq[16];

    const int tid = threadIdx.x;
    const int rowbase = blockIdx.x * 16;
    for (int idx = tid; idx < 128 * 128; idx += 256) {
        int k = idx >> 7, c = idx & 127;
        wT[c][k] = (bf16)W[idx];
    }
    if (tid < 16) { ssum[tid] = 0.f; ssq[tid] = 0.f; }
    __syncthreads();

    const int wave = tid >> 5, lane = tid & 31;
    const int lo = lane & 15, hi = lane >> 4;
    const int colbase = wave * 16;

    v8f acc = {};
    #pragma unroll
    for (int kk = 0; kk < 128; kk += 32) {
        const bf16* ap = A + (size_t)(rowbase + lo) * 128 + kk + hi * 16;
        v8bf alo = *(const v8bf*)ap;
        v8bf ahi = *(const v8bf*)(ap + 8);
        v16bf a;
        #pragma unroll
        for (int i = 0; i < 8; i++) { a[i] = alo[i]; a[i + 8] = ahi[i]; }
        v16bf b = *(const v16bf*)&wT[colbase + lo][kk + hi * 16];
        acc = wmma_bf16(a, b, acc);
    }
    const int col = colbase + lo;
    #pragma unroll
    for (int rr = 0; rr < 8; rr++) {
        int row = rowbase + rr + 8 * hi;
        zbuf[rr + 8 * hi][col] = acc[rr] + bias[col] + skip[(size_t)col * 2048 + row];
    }
    __syncthreads();

    const int r = tid >> 4, j = tid & 15;
    {
        float s = 0.f, s2 = 0.f;
        #pragma unroll
        for (int e = 0; e < 8; e++) {
            float v = zbuf[r][j * 8 + e];
            s += v; s2 += v * v;
        }
        atomicAdd(&ssum[r], s);
        atomicAdd(&ssq[r], s2);
    }
    __syncthreads();
    {
        float mean = ssum[r] * (1.0f / 128.0f);
        float var  = ssq[r] * (1.0f / 128.0f) - mean * mean;
        float rstd = rsqrtf(var + 1e-5f);
        #pragma unroll
        for (int e = 0; e < 8; e++) {
            int d = j * 8 + e;
            float v = (zbuf[r][d] - mean) * rstd * g[d] + beta[d];
            size_t o = (size_t)(rowbase + r) * 128 + d;
            z2b[o] = (bf16)v;
            z2f[o] = v;
        }
    }
}

// ---------------------------------------------------------------------------
// K4: h1 = gelu_exact(z2 @ w1 + b1). w1: [128][256]; block handles 16 rows x
// one 128-col half (blockIdx.x = rowtile*2 + half).
// ---------------------------------------------------------------------------
__global__ void __launch_bounds__(256)
ffn1_kernel(const bf16* __restrict__ A, const float* __restrict__ W1,
            const float* __restrict__ b1, bf16* __restrict__ h1) {
    __shared__ __align__(32) bf16 wT[128][128];

    const int tid = threadIdx.x;
    const int half = blockIdx.x & 1;
    const int rowbase = (blockIdx.x >> 1) * 16;
    for (int idx = tid; idx < 128 * 128; idx += 256) {
        int k = idx >> 7, c = idx & 127;
        wT[c][k] = (bf16)W1[(size_t)k * 256 + half * 128 + c];
    }
    __syncthreads();

    const int wave = tid >> 5, lane = tid & 31;
    const int lo = lane & 15, hi = lane >> 4;
    const int colbase = wave * 16;

    v8f acc = {};
    #pragma unroll
    for (int kk = 0; kk < 128; kk += 32) {
        const bf16* ap = A + (size_t)(rowbase + lo) * 128 + kk + hi * 16;
        v8bf alo = *(const v8bf*)ap;
        v8bf ahi = *(const v8bf*)(ap + 8);
        v16bf a;
        #pragma unroll
        for (int i = 0; i < 8; i++) { a[i] = alo[i]; a[i + 8] = ahi[i]; }
        v16bf b = *(const v16bf*)&wT[colbase + lo][kk + hi * 16];
        acc = wmma_bf16(a, b, acc);
    }
    const int col = colbase + lo;
    const int gcol = half * 128 + col;
    #pragma unroll
    for (int rr = 0; rr < 8; rr++) {
        int row = rowbase + rr + 8 * hi;
        float v = acc[rr] + b1[gcol];
        float gel = 0.5f * v * (1.0f + erff(v * 0.70710678118654752f));
        h1[(size_t)row * 256 + gcol] = (bf16)gel;
    }
}

// ---------------------------------------------------------------------------
// K5: out = LN_post(z2 + h1 @ w2 + b2), written transposed as [d][qpos].
// ---------------------------------------------------------------------------
__global__ void __launch_bounds__(256)
ffn2_kernel(const bf16* __restrict__ A, const float* __restrict__ W2,
            const float* __restrict__ b2, const float* __restrict__ z2f,
            const float* __restrict__ g, const float* __restrict__ beta,
            float* __restrict__ out) {
    __shared__ __align__(32) bf16 wT[128][256];   // wT[col][k], k=0..255
    __shared__ float zbuf[16][128];
    __shared__ float ssum[16], ssq[16];

    const int tid = threadIdx.x;
    const int rowbase = blockIdx.x * 16;
    for (int idx = tid; idx < 256 * 128; idx += 256) {
        int k = idx >> 7, c = idx & 127;
        wT[c][k] = (bf16)W2[(size_t)k * 128 + c];
    }
    if (tid < 16) { ssum[tid] = 0.f; ssq[tid] = 0.f; }
    __syncthreads();

    const int wave = tid >> 5, lane = tid & 31;
    const int lo = lane & 15, hi = lane >> 4;
    const int colbase = wave * 16;

    v8f acc = {};
    #pragma unroll
    for (int kk = 0; kk < 256; kk += 32) {
        const bf16* ap = A + (size_t)(rowbase + lo) * 256 + kk + hi * 16;
        v8bf alo = *(const v8bf*)ap;
        v8bf ahi = *(const v8bf*)(ap + 8);
        v16bf a;
        #pragma unroll
        for (int i = 0; i < 8; i++) { a[i] = alo[i]; a[i + 8] = ahi[i]; }
        v16bf b = *(const v16bf*)&wT[colbase + lo][kk + hi * 16];
        acc = wmma_bf16(a, b, acc);
    }
    const int col = colbase + lo;
    #pragma unroll
    for (int rr = 0; rr < 8; rr++) {
        int row = rowbase + rr + 8 * hi;
        zbuf[rr + 8 * hi][col] = acc[rr] + b2[col] + z2f[(size_t)row * 128 + col];
    }
    __syncthreads();

    const int r = tid >> 4, j = tid & 15;
    {
        float s = 0.f, s2 = 0.f;
        #pragma unroll
        for (int e = 0; e < 8; e++) {
            float v = zbuf[r][j * 8 + e];
            s += v; s2 += v * v;
        }
        atomicAdd(&ssum[r], s);
        atomicAdd(&ssq[r], s2);
    }
    __syncthreads();
    {
        float mean = ssum[r] * (1.0f / 128.0f);
        float var  = ssq[r] * (1.0f / 128.0f) - mean * mean;
        float rstd = rsqrtf(var + 1e-5f);
        #pragma unroll
        for (int e = 0; e < 8; e++) {
            int d = j * 8 + e;
            float v = (zbuf[r][d] - mean) * rstd * g[d] + beta[d];
            out[(size_t)d * 2048 + (rowbase + r)] = v;   // b Q d -> b d s...
        }
    }
}

// ---------------------------------------------------------------------------
extern "C" void kernel_launch(void* const* d_in, const int* in_sizes, int n_in,
                              void* d_out, int out_size, void* d_ws, size_t ws_size,
                              hipStream_t stream) {
    const float* q     = (const float*)d_in[0];
    const float* k     = (const float*)d_in[1];
    const float* v     = (const float*)d_in[2];
    const float* skip  = (const float*)d_in[3];
    // d_in[4]=heads, d_in[5]=dim_head (scalars; hardcoded 4/32)
    const float* lnq_g = (const float*)d_in[6];
    const float* lnq_b = (const float*)d_in[7];
    const float* wq    = (const float*)d_in[8];
    const float* bq    = (const float*)d_in[9];
    const float* lnk_g = (const float*)d_in[10];
    const float* lnk_b = (const float*)d_in[11];
    const float* wk    = (const float*)d_in[12];
    const float* bk    = (const float*)d_in[13];
    const float* lnv_g = (const float*)d_in[14];
    const float* lnv_b = (const float*)d_in[15];
    const float* wv    = (const float*)d_in[16];
    const float* bv    = (const float*)d_in[17];
    const float* w_o   = (const float*)d_in[18];
    const float* b_o   = (const float*)d_in[19];
    const float* pre_g = (const float*)d_in[20];
    const float* pre_b = (const float*)d_in[21];
    const float* w1    = (const float*)d_in[22];
    const float* b1    = (const float*)d_in[23];
    const float* w2    = (const float*)d_in[24];
    const float* b2    = (const float*)d_in[25];
    const float* post_g= (const float*)d_in[26];
    const float* post_b= (const float*)d_in[27];

    char* w = (char*)d_ws;
    bf16*  qp  = (bf16*)(w);                                  // 8192*128 bf16 = 2 MB
    bf16*  kp  = (bf16*)(w + 2097152);                        // 2 MB
    bf16*  vpT = (bf16*)(w + 4194304);                        // 128 x 8192 bf16 = 2 MB
    bf16*  aat = (bf16*)(w + 6291456);                        // 2048*128 bf16 = 512 KB
    bf16*  z2b = (bf16*)(w + 6291456 + 524288);               // 512 KB
    float* z2f = (float*)(w + 6291456 + 1048576);             // 1 MB
    bf16*  h1  = (bf16*)(w + 6291456 + 2097152);              // 2048*256 bf16 = 1 MB

    // 1/sqrt(32) * log2(e): scores carry log2(e) so softmax uses v_exp (base 2)
    const float scale_q = 0.17677669529663688f * 1.44269504088896340f;

    ln_proj_kernel<0><<<512, 256, 0, stream>>>(q, lnq_g, lnq_b, wq, bq, qp, scale_q);
    ln_proj_kernel<0><<<512, 256, 0, stream>>>(k, lnk_g, lnk_b, wk, bk, kp, 1.0f);
    ln_proj_kernel<1><<<512, 256, 0, stream>>>(v, lnv_g, lnv_b, wv, bv, vpT, 1.0f);
    attn_kernel<<<128, 128, 0, stream>>>(qp, kp, vpT, aat);
    out_proj_kernel<<<128, 256, 0, stream>>>(aat, w_o, b_o, skip, pre_g, pre_b, z2b, z2f);
    ffn1_kernel<<<256, 256, 0, stream>>>(z2b, w1, b1, h1);
    ffn2_kernel<<<128, 256, 0, stream>>>(h1, w2, b2, z2f, post_g, post_b, (float*)d_out);
}
